// Decoder_75170517615051
// MI455X (gfx1250) — matrix operand order repaired and checked by reading, not compiled, not run on hardware
//
#include <hip/hip_runtime.h>
#include <hip/hip_bf16.h>

typedef __bf16 bf16_t;
typedef __attribute__((ext_vector_type(16))) __bf16 v16bf;
typedef __attribute__((ext_vector_type(8)))  __bf16 v8bf;
typedef __attribute__((ext_vector_type(8)))  float  v8f;

#define NB 64      // batch
#define NP 196     // image positions
#define NC 2048    // channels
#define ND 512     // hidden
#define NE 512     // embed
#define NV 30000   // vocab
#define NL 32      // max text len
#define NT 31      // steps

// ---------------------------------------------------------------------------
// Register-blocked bf16 WMMA GEMM:
//   D[M,N] = A[M,K](bf16,row-major) * Bt[N,K](bf16)^T + bias[N]
// Each wave computes a (BM*16) x (BN*16) output block: BM*BN accumulators,
// A tiles loaded once per K-step and reused across BN, B across BM.
// Tail tiles clamp loads to tile 0 (in-bounds) and guard stores only, so
// EXEC stays all-ones for every WMMA. 8 waves per block.
// ---------------------------------------------------------------------------
template<int BM, int BN>
__global__ __launch_bounds__(256)
void wmma_gemm_bf16(const bf16_t* __restrict__ A, const bf16_t* __restrict__ Bt,
                    const float* __restrict__ bias, float* __restrict__ D,
                    int M, int N, int K, long ldD,
                    const int* __restrict__ tlen, int tstep)
{
    const int lane  = threadIdx.x & 31;
    const int wave  = threadIdx.x >> 5;
    const int tilesM = M >> 4, tilesN = N >> 4;
    const int gN = (tilesN + BN - 1) / BN;
    const int gM = (tilesM + BM - 1) / BM;
    const int g  = blockIdx.x * 8 + wave;
    if (g >= gM * gN) return;                         // uniform per wave
    const int mg = g / gN, ng = g % gN;
    const int half = lane >> 4;                       // 0 | 1
    const int l16  = lane & 15;
    const int kA   = half * 16;                       // A: e0-15 -> K+kA..+15
    const int kB   = half * 16;                       // B: e0-15 -> K+kB..+15

    const bf16_t* aptr[BM];
    const bf16_t* bptr[BN];
    int  mrow[BM], ncol[BN];
    bool mv[BM], nv[BN];
#pragma unroll
    for (int i = 0; i < BM; ++i) {
        int mt = mg * BM + i;
        mv[i] = mt < tilesM;
        mrow[i] = (mv[i] ? mt : 0) * 16;
        aptr[i] = A + (long)(mrow[i] + l16) * K;
    }
#pragma unroll
    for (int j = 0; j < BN; ++j) {
        int nt = ng * BN + j;
        nv[j] = nt < tilesN;
        ncol[j] = (nv[j] ? nt : 0) * 16;
        bptr[j] = Bt + (long)(ncol[j] + l16) * K;
    }

    v8f acc[BM][BN];
#pragma unroll
    for (int i = 0; i < BM; ++i)
#pragma unroll
        for (int j = 0; j < BN; ++j)
            acc[i][j] = (v8f){};

    for (int kt = 0; kt < K; kt += 32) {
        v16bf av[BM], bv[BN];
#pragma unroll
        for (int i = 0; i < BM; ++i) {
            v8bf a0 = *(const v8bf*)(aptr[i] + kt + kA);
            v8bf a1 = *(const v8bf*)(aptr[i] + kt + kA + 8);
            av[i] = __builtin_shufflevector(a0, a1, 0,1,2,3,4,5,6,7,8,9,10,11,12,13,14,15);
        }
#pragma unroll
        for (int j = 0; j < BN; ++j)
            bv[j] = *(const v16bf*)(bptr[j] + kt + kB);
#pragma unroll
        for (int i = 0; i < BM; ++i)
#pragma unroll
            for (int j = 0; j < BN; ++j)
                acc[i][j] = __builtin_amdgcn_wmma_f32_16x16x32_bf16(
                    false, av[i], false, bv[j], (short)0, acc[i][j], false, false);
    }

#pragma unroll
    for (int i = 0; i < BM; ++i) {
        if (!mv[i]) continue;
        const int orow = mrow[i] + half * 8;
#pragma unroll
        for (int j = 0; j < BN; ++j) {
            if (!nv[j]) continue;
            const int bcol = ncol[j] + l16;
            const float bvl = bias ? bias[bcol] : 0.0f;
#pragma unroll
            for (int e = 0; e < 8; ++e) {
                int r = orow + e;
                float v = acc[i][j][e] + bvl;
                if (tlen && !(tstep < tlen[r])) v = 0.0f;
                D[(long)r * ldD + bcol] = v;
            }
        }
    }
}

// ---------------------------------------------------------------------------
// Weight convert + transpose: out[n*ldOut + kOff + k] = bf16(in[k*N + n])
// ---------------------------------------------------------------------------
__global__ void conv_transpose_bf16(const float* __restrict__ in, bf16_t* __restrict__ out,
                                    int K, int N, int ldOut, int kOff)
{
    long id = (long)blockIdx.x * 256 + threadIdx.x;
    if (id >= (long)K * N) return;
    int n = (int)(id / K), k = (int)(id % K);
    out[(long)n * ldOut + kOff + k] = (bf16_t)in[(long)k * N + n];
}

__global__ void conv_bf16(const float* __restrict__ in, bf16_t* __restrict__ out, long n)
{
    long id = (long)blockIdx.x * 256 + threadIdx.x;
    if (id < n) out[id] = (bf16_t)in[id];
}

__global__ void copy_f32(const float* __restrict__ a, float* __restrict__ o, int n)
{
    int i = blockIdx.x * 256 + threadIdx.x;
    if (i < n) o[i] = a[i];
}

__global__ void add_f32(const float* __restrict__ a, const float* __restrict__ b,
                        float* __restrict__ o, int n)
{
    int i = blockIdx.x * 256 + threadIdx.x;
    if (i < n) o[i] = a[i] + b[i];
}

// ---------------------------------------------------------------------------
// Stable argsort of 64 lengths, descending.
// ---------------------------------------------------------------------------
__global__ void sort_kernel(const int* __restrict__ tl, int* __restrict__ idx,
                            int* __restrict__ tlen)
{
    __shared__ int s[NB];
    int i = threadIdx.x;
    s[i] = tl[i];
    __syncthreads();
    int ti = s[i], r = 0;
    for (int j = 0; j < NB; ++j) {
        int tj = s[j];
        r += (tj > ti) || (tj == ti && j < i);
    }
    idx[r] = i;
    tlen[r] = ti - 1;
}

__global__ void write_misc(const int* __restrict__ text, const int* __restrict__ tl,
                           const int* __restrict__ idx, float* __restrict__ out_text,
                           float* __restrict__ out_tlen, float* __restrict__ out_idx)
{
    int b = blockIdx.x;
    int src = idx[b];
    for (int l = threadIdx.x; l < NL; l += 32)
        out_text[b * NL + l] = (float)text[src * NL + l];
    if (threadIdx.x == 0) {
        out_tlen[b] = (float)(tl[src] - 1);
        out_idx[b]  = (float)src;
    }
}

__global__ void img_mean_kernel(const float* __restrict__ img, float* __restrict__ mean)
{
    int b = blockIdx.y;
    int c = blockIdx.x * 256 + threadIdx.x;            // 2048 = 8 * 256
    const float* p = img + (long)b * NP * NC + c;
    float s = 0.f;
    for (int pp = 0; pp < NP; ++pp) s += p[(long)pp * NC];
    mean[b * NC + c] = s * (1.0f / (float)NP);
}

__global__ void split_hc(const float* __restrict__ hc, float* __restrict__ h,
                         float* __restrict__ c, bf16_t* __restrict__ hb)
{
    int b = blockIdx.x, d = threadIdx.x;               // 512 threads
    float hv = hc[b * 2 * ND + d];
    float cv = hc[b * 2 * ND + ND + d];
    h[b * ND + d] = hv;
    c[b * ND + d] = cv;
    hb[b * ND + d] = (bf16_t)hv;
}

__global__ void img_sorted_bf16(const float* __restrict__ img, const int* __restrict__ idx,
                                bf16_t* __restrict__ out)
{
    long id = (long)blockIdx.x * 256 + threadIdx.x;
    const long per = (long)NP * NC;
    if (id >= (long)NB * per) return;
    int b = (int)(id / per);
    long rem = id % per;
    out[id] = (bf16_t)img[(long)idx[b] * per + rem];
}

__global__ void embed_gather(const float* __restrict__ table, const int* __restrict__ text,
                             const int* __restrict__ idx, bf16_t* __restrict__ out)
{
    int bt = blockIdx.x;                               // b*NT + t
    int b = bt / NT, t = bt % NT;
    int tok = text[idx[b] * NL + t];
    const float* src = table + (long)tok * NE;
    bf16_t* dst = out + (long)bt * NE;
    for (int e = threadIdx.x; e < NE; e += 256) dst[e] = (bf16_t)src[e];
}

// ---------------------------------------------------------------------------
// Fused attention: score=relu(en_attn+de)@W_at+b_at; softmax over P; ctx.
// One block per batch row.
// ---------------------------------------------------------------------------
__global__ __launch_bounds__(256)
void attention_kernel(const float* __restrict__ en_attn, const float* __restrict__ dg,
                      const float* __restrict__ W_at, const float* __restrict__ b_at,
                      const float* __restrict__ img, const int* __restrict__ idx,
                      const int* __restrict__ tlen, int t,
                      float* __restrict__ ctx, float* __restrict__ attn_out)
{
    __shared__ float de_s[ND], wat_s[ND], sc_s[NP + 4], red[256];
    int b = blockIdx.x, tid = threadIdx.x;
    for (int a = tid; a < ND; a += 256) {
        de_s[a]  = dg[b * 2560 + a];
        wat_s[a] = W_at[a];
    }
    __syncthreads();
    int lane = tid & 31, grp = tid >> 5;
    const float* ea = en_attn + (long)b * NP * ND;
    for (int pb = 0; pb < NP; pb += 8) {
        int p = pb + grp;
        float s = 0.f;
        if (p < NP) {
            const float* row = ea + (long)p * ND;
            for (int a = lane; a < ND; a += 32) {
                float v = row[a] + de_s[a];
                s += (v > 0.f ? v : 0.f) * wat_s[a];
            }
        }
        for (int m = 16; m; m >>= 1) s += __shfl_xor(s, m);
        if (p < NP && lane == 0) sc_s[p] = s + b_at[0];
    }
    __syncthreads();
    // softmax over NP
    float mx = -1e30f;
    for (int p = tid; p < NP; p += 256) mx = fmaxf(mx, sc_s[p]);
    red[tid] = mx; __syncthreads();
    for (int st = 128; st; st >>= 1) { if (tid < st) red[tid] = fmaxf(red[tid], red[tid + st]); __syncthreads(); }
    mx = red[0]; __syncthreads();
    float sum = 0.f;
    for (int p = tid; p < NP; p += 256) { float e = __expf(sc_s[p] - mx); sc_s[p] = e; sum += e; }
    red[tid] = sum; __syncthreads();
    for (int st = 128; st; st >>= 1) { if (tid < st) red[tid] += red[tid + st]; __syncthreads(); }
    float inv = 1.0f / red[0];
    __syncthreads();
    for (int p = tid; p < NP; p += 256) sc_s[p] *= inv;
    __syncthreads();
    bool m = t < tlen[b];
    for (int p = tid; p < NP; p += 256)
        attn_out[((long)b * NT + t) * NP + p] = m ? sc_s[p] : 0.f;
    const float* im = img + (long)idx[b] * NP * NC;
    for (int c = tid; c < NC; c += 256) {
        float a = 0.f;
        for (int p = 0; p < NP; ++p) a += sc_s[p] * im[(long)p * NC + c];
        ctx[b * NC + c] = a;
    }
}

// xh = [ctx * sigmoid(gatelin) (2048) | emb_t (512) | h (512)]  in bf16
__global__ void assemble_xh(const float* __restrict__ ctx, const float* __restrict__ dg,
                            const bf16_t* __restrict__ embb, const bf16_t* __restrict__ hb,
                            bf16_t* __restrict__ xh, int t)
{
    int b = blockIdx.x;
    for (int c = threadIdx.x; c < 3072; c += 256) {
        bf16_t v;
        if (c < NC) {
            float g = dg[b * 2560 + ND + c];
            g = 1.0f / (1.0f + __expf(-g));
            v = (bf16_t)(ctx[b * NC + c] * g);
        } else if (c < NC + NE) {
            v = embb[((long)b * NT + t) * NE + (c - NC)];
        } else {
            v = hb[b * ND + (c - NC - NE)];
        }
        xh[(long)b * 3072 + c] = v;
    }
}

__global__ void lstm_cell(const float* __restrict__ gates, float* __restrict__ h,
                          float* __restrict__ c, bf16_t* __restrict__ hb,
                          bf16_t* __restrict__ hnb, const int* __restrict__ tlen, int t)
{
    int b = blockIdx.x, d = threadIdx.x;               // 512 threads
    float gi = gates[b * 2048 + d];
    float gf = gates[b * 2048 + 512 + d];
    float gg = gates[b * 2048 + 1024 + d];
    float go = gates[b * 2048 + 1536 + d];
    float si = 1.f / (1.f + __expf(-gi));
    float sf = 1.f / (1.f + __expf(-gf));
    float so = 1.f / (1.f + __expf(-go));
    float cn = sf * c[b * ND + d] + si * tanhf(gg);
    float hn = so * tanhf(cn);
    hnb[b * ND + d] = (bf16_t)hn;
    if (t < tlen[b]) { h[b * ND + d] = hn; c[b * ND + d] = cn; }
    hb[b * ND + d] = (bf16_t)h[b * ND + d];
}

// ---------------------------------------------------------------------------
extern "C" void kernel_launch(void* const* d_in, const int* in_sizes, int n_in,
                              void* d_out, int out_size, void* d_ws, size_t ws_size,
                              hipStream_t stream)
{
    const float* encoder_out = (const float*)d_in[0];
    const int*   text        = (const int*)d_in[1];
    const int*   text_len    = (const int*)d_in[2];
    const float* W_init_h = (const float*)d_in[3];  const float* b_init_h = (const float*)d_in[4];
    const float* W_init_c = (const float*)d_in[5];  const float* b_init_c = (const float*)d_in[6];
    const float* W_en     = (const float*)d_in[7];  const float* b_en     = (const float*)d_in[8];
    const float* W_de     = (const float*)d_in[9];  const float* b_de     = (const float*)d_in[10];
    const float* W_at     = (const float*)d_in[11]; const float* b_at     = (const float*)d_in[12];
    const float* W_gate   = (const float*)d_in[13]; const float* b_gate   = (const float*)d_in[14];
    const float* W_ih     = (const float*)d_in[15]; const float* b_ih     = (const float*)d_in[16];
    const float* W_hh     = (const float*)d_in[17]; const float* b_hh     = (const float*)d_in[18];
    const float* embed    = (const float*)d_in[19];
    const float* W_head   = (const float*)d_in[20]; const float* b_head   = (const float*)d_in[21];

    // ---- output layout (flat, return order) ----
    float* outf     = (float*)d_out;
    float* out_pred = outf;                                     // NB*NT*NV
    float* out_text = out_pred + (size_t)NB * NT * NV;          // NB*NL
    float* out_tlen = out_text + (size_t)NB * NL;               // NB
    float* out_attn = out_tlen + NB;                            // NB*NT*NP
    float* out_idx  = out_attn + (size_t)NB * NT * NP;          // NB

    // ---- workspace carving ----
    char* wp = (char*)d_ws;
    auto carve = [&](size_t bytes) -> void* {
        void* r = (void*)wp;
        wp += (bytes + 255) & ~(size_t)255;
        return r;
    };
    bf16_t* imgb    = (bf16_t*)carve((size_t)NB * NP * NC * 2); // sorted image, bf16
    float*  en_attn = (float*) carve((size_t)NB * NP * ND * 4);
    bf16_t* Whead_t = (bf16_t*)carve((size_t)NV * ND * 2);
    bf16_t* Wg_t    = (bf16_t*)carve((size_t)2048 * 3072 * 2);  // [W_ih;W_hh] N=2048,K=3072
    bf16_t* Wdg_t   = (bf16_t*)carve((size_t)2560 * ND * 2);    // [W_de|W_gate] N=2560,K=512
    bf16_t* Wen_t   = (bf16_t*)carve((size_t)ND * NC * 2);
    bf16_t* Winit_t = (bf16_t*)carve((size_t)1024 * NC * 2);    // [W_init_h|W_init_c]
    bf16_t* embb    = (bf16_t*)carve((size_t)NB * NT * NE * 2);
    float*  bias_dg = (float*) carve(2560 * 4);
    float*  bias_g  = (float*) carve(2048 * 4);
    float*  bias_in = (float*) carve(1024 * 4);
    float*  imean   = (float*) carve((size_t)NB * NC * 4);
    bf16_t* imeanb  = (bf16_t*)carve((size_t)NB * NC * 2);
    float*  hc0     = (float*) carve((size_t)NB * 1024 * 4);
    float*  hst     = (float*) carve((size_t)NB * ND * 4);
    float*  cst     = (float*) carve((size_t)NB * ND * 4);
    bf16_t* hb      = (bf16_t*)carve((size_t)NB * ND * 2);
    bf16_t* hnb     = (bf16_t*)carve((size_t)NB * ND * 2);
    float*  dg      = (float*) carve((size_t)NB * 2560 * 4);
    float*  ctx     = (float*) carve((size_t)NB * NC * 4);
    bf16_t* xh      = (bf16_t*)carve((size_t)NB * 3072 * 2);
    float*  gates   = (float*) carve((size_t)NB * 2048 * 4);
    int*    idx     = (int*)   carve(NB * 4);
    int*    tlen    = (int*)   carve(NB * 4);

    // big GEMMs: 4x4 register blocking (32 FLOP/B from L2)
    auto gemm44 = [&](const bf16_t* A, const bf16_t* Bt, const float* bias, float* Dst,
                      int M, int N, int K, long ldD, const int* tl, int t) {
        int groups = ((M / 16 + 3) / 4) * ((N / 16 + 3) / 4);
        wmma_gemm_bf16<4, 4><<<(groups + 7) / 8, 256, 0, stream>>>(
            A, Bt, bias, Dst, M, N, K, ldD, tl, t);
    };
    // small latency-bound GEMMs: 2x2 blocking, more waves in flight
    auto gemm22 = [&](const bf16_t* A, const bf16_t* Bt, const float* bias, float* Dst,
                      int M, int N, int K, long ldD, const int* tl, int t) {
        int groups = ((M / 16 + 1) / 2) * ((N / 16 + 1) / 2);
        wmma_gemm_bf16<2, 2><<<(groups + 7) / 8, 256, 0, stream>>>(
            A, Bt, bias, Dst, M, N, K, ldD, tl, t);
    };
    auto trans = [&](const float* in, bf16_t* out, int K, int N, int ldOut, int kOff) {
        long n = (long)K * N;
        conv_transpose_bf16<<<(int)((n + 255) / 256), 256, 0, stream>>>(in, out, K, N, ldOut, kOff);
    };

    // ---- prep ----
    sort_kernel<<<1, NB, 0, stream>>>(text_len, idx, tlen);
    write_misc<<<NB, 32, 0, stream>>>(text, text_len, idx, out_text, out_tlen, out_idx);

    trans(W_en,     Wen_t,                 NC, ND,   NC, 0);
    trans(W_init_h, Winit_t,               NC, ND,   NC, 0);
    trans(W_init_c, Winit_t + 512 * NC,    NC, ND,   NC, 0);
    trans(W_de,     Wdg_t,                 ND, ND,   ND, 0);
    trans(W_gate,   Wdg_t + 512 * ND,      ND, NC,   ND, 0);
    trans(W_ih,     Wg_t,                  2560, 2048, 3072, 0);
    trans(W_hh,     Wg_t,                  ND,   2048, 3072, 2560);
    trans(W_head,   Whead_t,               ND, NV,   ND, 0);

    copy_f32<<<2, 256, 0, stream>>>(b_de, bias_dg, 512);
    copy_f32<<<8, 256, 0, stream>>>(b_gate, bias_dg + 512, 2048);
    add_f32<<<8, 256, 0, stream>>>(b_ih, b_hh, bias_g, 2048);
    copy_f32<<<2, 256, 0, stream>>>(b_init_h, bias_in, 512);
    copy_f32<<<2, 256, 0, stream>>>(b_init_c, bias_in + 512, 512);

    img_mean_kernel<<<dim3(NC / 256, NB), 256, 0, stream>>>(encoder_out, imean);
    conv_bf16<<<(NB * NC + 255) / 256, 256, 0, stream>>>(imean, imeanb, (long)NB * NC);
    gemm22(imeanb, Winit_t, bias_in, hc0, NB, 1024, NC, 1024, nullptr, 0);  // h0|c0
    split_hc<<<NB, ND, 0, stream>>>(hc0, hst, cst, hb);

    {   // sorted image in bf16, then en_attn = img_s @ W_en + b_en
        long n = (long)NB * NP * NC;
        img_sorted_bf16<<<(int)((n + 255) / 256), 256, 0, stream>>>(encoder_out, idx, imgb);
    }
    gemm44(imgb, Wen_t, b_en, en_attn, NB * NP, ND, NC, ND, nullptr, 0);
    embed_gather<<<NB * NT, 256, 0, stream>>>(embed, text, idx, embb);

    // ---- recurrent steps ----
    for (int t = 0; t < NT; ++t) {
        gemm22(hb, Wdg_t, bias_dg, dg, NB, 2560, ND, 2560, nullptr, 0);    // de | gatelin
        attention_kernel<<<NB, 256, 0, stream>>>(en_attn, dg, W_at, b_at,
                                                 encoder_out, idx, tlen, t, ctx, out_attn);
        assemble_xh<<<NB, 256, 0, stream>>>(ctx, dg, embb, hb, xh, t);
        gemm22(xh, Wg_t, bias_g, gates, NB, 2048, 3072, 2048, nullptr, 0);
        lstm_cell<<<NB, ND, 0, stream>>>(gates, hst, cst, hb, hnb, tlen, t);
        gemm44(hnb, Whead_t, b_head, out_pred + (long)t * NV,
               NB, NV, ND, (long)NT * NV, tlen, t);                         // masked head
    }
}
